// SyntaxPositionalEmbedding_78331613544613
// MI455X (gfx1250) — matrix operand order repaired and checked by reading, not compiled, not run on hardware
//
#include <hip/hip_runtime.h>
#include <hip/hip_bf16.h>
#include <cstdint>

// ---- problem dims ----
#define B_   16
#define S_   2048
#define H_   768
#define M_   (B_ * (S_ - 1))   // 32752 = 2047 * 16
#define DDIM 128
#define CDIM 128
#define MIX  128

typedef __bf16 bf16_t;
typedef bf16_t v16bf __attribute__((ext_vector_type(16)));
typedef float  v8f   __attribute__((ext_vector_type(8)));

__device__ __forceinline__ uint16_t f2bf(float f) {
    // round-to-nearest-even f32 -> bf16
    uint32_t x = __float_as_uint(f);
    x += 0x7FFFu + ((x >> 16) & 1u);
    return (uint16_t)(x >> 16);
}

__device__ __forceinline__ uint2 pack4bf(float4 v) {
    uint2 p;
    p.x = (uint32_t)f2bf(v.x) | ((uint32_t)f2bf(v.y) << 16);
    p.y = (uint32_t)f2bf(v.z) | ((uint32_t)f2bf(v.w) << 16);
    return p;
}

union FragU { uint4 u[2]; v16bf v; };

// -------------------------------------------------------------------------
// f32 -> bf16 bulk convert, 4 elems/thread (n must be multiple of 4)
// -------------------------------------------------------------------------
__global__ void cvt_bf16_kernel(const float* __restrict__ in,
                                uint16_t* __restrict__ out, int n) {
    int i = (blockIdx.x * blockDim.x + threadIdx.x) * 4;
    if (i < n) {
        float4 v = *(const float4*)(in + i);
        *(uint2*)(out + i) = pack4bf(v);
    }
}

// -------------------------------------------------------------------------
// seqs_u = seqs + E_u[u]; f32 to d_out and packed bf16 copy for GEMM2 A
// grid = B*S blocks, 192 threads (one float4 per thread)
// -------------------------------------------------------------------------
__global__ void embed_add_kernel(const float* __restrict__ seqs,
                                 const int* __restrict__ u,
                                 const float* __restrict__ Eu,
                                 float* __restrict__ outF,
                                 uint16_t* __restrict__ seqsbf) {
    size_t row = blockIdx.x;
    const float* er = Eu + (size_t)u[row] * H_;
    int e = threadIdx.x * 4;
    size_t base = row * H_ + e;
    float4 sv = *(const float4*)(seqs + base);
    float4 ev = *(const float4*)(er + e);
    float4 v = make_float4(sv.x + ev.x, sv.y + ev.y, sv.z + ev.z, sv.w + ev.w);
    *(float4*)(outF + base) = v;
    *(uint2*)(seqsbf + base) = pack4bf(v);
}

// -------------------------------------------------------------------------
// A1[m, 0:128] = bf16(E_d[d[m]]);  A1[m, 128:256] = bf16(E_c[c[m]])
// grid = M/4 blocks, 256 threads = 4 rows x 64 lanes (float4 each)
// -------------------------------------------------------------------------
__global__ void gather_dc_kernel(const int* __restrict__ d,
                                 const int* __restrict__ c,
                                 const float* __restrict__ Ed,
                                 const float* __restrict__ Ec,
                                 uint16_t* __restrict__ A1) {
    int m = blockIdx.x * 4 + (threadIdx.x >> 6);
    int e = (threadIdx.x & 63) * 4;
    const float* src = (e < DDIM) ? Ed + (size_t)d[m] * DDIM + e
                                  : Ec + (size_t)c[m] * CDIM + (e - DDIM);
    float4 v = *(const float4*)src;
    *(uint2*)(A1 + (size_t)m * (DDIM + CDIM) + e) = pack4bf(v);
}

// -------------------------------------------------------------------------
// WMMA fragment loads.
// A fragment: 16 bf16/lane = two 16B chunks.
//   lane holds K = kk + hi*16 + [0..15] (two contiguous 16B chunks).
// B fragment: lane holds K = kk + hi*16 + [0..15], contiguous 32B.
// MODE != 0: two-source concat view; chunks are 8-aligned and the concat
// boundary (768) is 8-aligned, so each chunk lies fully in one source.
// -------------------------------------------------------------------------
template<int MODE>
__device__ __forceinline__ void load_a(const uint16_t* __restrict__ aRow0,
                                       const uint16_t* __restrict__ aRow1,
                                       int kk, int hi, FragU& a) {
    int c0 = kk + hi * 16;
    int c1 = c0 + 8;
    if (MODE == 0) {
        a.u[0] = *(const uint4*)(aRow0 + c0);
        a.u[1] = *(const uint4*)(aRow0 + c1);
    } else {
        a.u[0] = (c0 < H_) ? *(const uint4*)(aRow0 + c0)
                           : *(const uint4*)(aRow1 + (c0 - H_));
        a.u[1] = (c1 < H_) ? *(const uint4*)(aRow0 + c1)
                           : *(const uint4*)(aRow1 + (c1 - H_));
    }
}

__device__ __forceinline__ void load_b(const uint16_t* const* wRow,
                                       int kk, int hi, FragU* b) {
    int kb = kk + hi * 16;
#pragma unroll
    for (int j = 0; j < 4; ++j) {
        b[j].u[0] = *(const uint4*)(wRow[j] + kb);
        b[j].u[1] = *(const uint4*)(wRow[j] + kb + 8);
    }
}

__device__ __forceinline__ void mma8(const FragU& aA, const FragU& aB,
                                     const FragU* b, v8f* accA, v8f* accB) {
#pragma unroll
    for (int j = 0; j < 4; ++j) {
        accA[j] = __builtin_amdgcn_wmma_f32_16x16x32_bf16(
            false, aA.v, false, b[j].v, (short)0, accA[j], false, false);
        accB[j] = __builtin_amdgcn_wmma_f32_16x16x32_bf16(
            false, aB.v, false, b[j].v, (short)0, accB[j], false, false);
    }
}

// -------------------------------------------------------------------------
// Tiled bf16 WMMA GEMM with ReLU:  out[M, NDIM] = relu(A[M, KDIM] @ W^T)
// W is [NDIM, KDIM] bf16 row-major (== B-fragment layout, contiguous loads).
// MODE 0: A rows contiguous at A0 (lda = KDIM)
// MODE 1: pair-concat view of seqs_u bf16: m = b*2047+s -> global row m + b,
//         k < 768 from row, k >= 768 from row+1  (KDIM = 1536)
// MODE 2: concat(ph[M,768], dc[M,128])           (KDIM = 896)
// One wave computes a 32x64 tile (m-tiles p and p+1024, second clamped):
// the 4 B fragments are shared by 2 A fragments -> 8 WMMAs per 12 b128 loads.
// K loop double-buffered in registers so phase k+1 fetches overlap phase k math.
// -------------------------------------------------------------------------
template<int MODE, int KDIM, int NDIM, bool OUTF32>
__global__ __launch_bounds__(256)
void gemm_relu_wmma(const uint16_t* __restrict__ A0,
                    const uint16_t* __restrict__ A1,
                    const uint16_t* __restrict__ W,
                    float* __restrict__ outF,
                    uint16_t* __restrict__ outB) {
    static_assert(KDIM % 64 == 0, "two-phase pipeline needs K % 64 == 0");
    constexpr int NCH   = NDIM / 64;
    constexpr int MTOT  = M_ / 16;            // 2047 m-tiles
    constexpr int MT2   = (MTOT + 1) / 2;     // 1024 tile pairs
    constexpr int TASKS = MT2 * NCH;

    int wave = blockIdx.x * 8 + (threadIdx.x >> 5);
    if (wave >= TASKS) return;                // uniform per wave: EXEC all-ones
    int p      = wave / NCH;
    int nChunk = wave - p * NCH;
    int mTileA = p;
    int mTileB = p + MT2;                     // clamp last pair (dup compute,
    if (mTileB > MTOT - 1) mTileB = MTOT - 1; //  identical values -> benign)

    int lane = threadIdx.x & 31;
    int hi   = lane >> 4;                     // half-wave select
    int lo   = lane & 15;

    // Per-lane A row pointers for both m-tiles
    const uint16_t* aR0[2];
    const uint16_t* aR1[2];
    int mIdx[2] = {mTileA * 16 + lo, mTileB * 16 + lo};
#pragma unroll
    for (int t = 0; t < 2; ++t) {
        int m = mIdx[t];
        if (MODE == 0) {
            aR0[t] = A0 + (size_t)m * KDIM;
            aR1[t] = aR0[t];
        } else if (MODE == 1) {
            int b = m / (S_ - 1);             // batch index
            size_t rg = (size_t)(m + b);      // global row in seqs_u
            aR0[t] = A0 + rg * H_;            // k in [0,768)
            aR1[t] = A0 + (rg + 1) * H_;      // k-768
        } else {
            aR0[t] = A0 + (size_t)m * H_;     // ph row,  k in [0,768)
            aR1[t] = A1 + (size_t)m * MIX;    // dc row,  k-768
        }
    }

    // Per-lane B row pointers: lane lo is output column, contiguous K
    const uint16_t* wRow[4];
#pragma unroll
    for (int j = 0; j < 4; ++j)
        wRow[j] = W + (size_t)(nChunk * 64 + j * 16 + lo) * KDIM;

    v8f accA[4] = {}, accB[4] = {};
    FragU aA0, aB0, aA1, aB1;
    FragU b0[4], b1[4];

    load_a<MODE>(aR0[0], aR1[0], 0, hi, aA0);
    load_a<MODE>(aR0[1], aR1[1], 0, hi, aB0);
    load_b(wRow, 0, hi, b0);
#pragma unroll 1
    for (int kk = 0; kk < KDIM; kk += 64) {
        load_a<MODE>(aR0[0], aR1[0], kk + 32, hi, aA1);
        load_a<MODE>(aR0[1], aR1[1], kk + 32, hi, aB1);
        load_b(wRow, kk + 32, hi, b1);
        mma8(aA0, aB0, b0, accA, accB);
        if (kk + 64 < KDIM) {
            load_a<MODE>(aR0[0], aR1[0], kk + 64, hi, aA0);
            load_a<MODE>(aR0[1], aR1[1], kk + 64, hi, aB0);
            load_b(wRow, kk + 64, hi, b0);
        }
        mma8(aA1, aB1, b1, accA, accB);
    }

    // C/D layout: VGPR r, lanes 0-15 -> (row = 16*mTile + r,   col = lo)
    //                     lanes 16-31 -> (row = 16*mTile + 8+r, col = lo)
#pragma unroll
    for (int j = 0; j < 4; ++j) {
        int col = nChunk * 64 + j * 16 + lo;
#pragma unroll
        for (int r = 0; r < 8; ++r) {
            int rowA = mTileA * 16 + hi * 8 + r;
            int rowB = mTileB * 16 + hi * 8 + r;
            float vA = fmaxf(accA[j][r], 0.0f);
            float vB = fmaxf(accB[j][r], 0.0f);
            if (OUTF32) {
                outF[(size_t)rowA * NDIM + col] = vA;
                outF[(size_t)rowB * NDIM + col] = vB;
            } else {
                outB[(size_t)rowA * NDIM + col] = f2bf(vA);
                outB[(size_t)rowB * NDIM + col] = f2bf(vB);
            }
        }
    }
}

// -------------------------------------------------------------------------
// out[b,s] += (s <= S-2 ? m[b,s] : 0) + (s >= 2 ? m[b,s-1] : 0)
// grid = B*S blocks, 192 threads (one float4 per thread)
// -------------------------------------------------------------------------
__global__ void scatter_add_kernel(const float* __restrict__ mF,
                                   float* __restrict__ outF) {
    int row = blockIdx.x;
    int b = row >> 11;          // /2048
    int s = row & (S_ - 1);
    bool has0 = (s <= S_ - 2);
    bool has1 = (s >= 2);
    int e = threadIdx.x * 4;
    size_t base = (size_t)row * H_ + e;
    float4 acc = *(const float4*)(outF + base);
    if (has0) {
        float4 v = *(const float4*)(mF + ((size_t)(b * (S_ - 1) + s)) * H_ + e);
        acc.x += v.x; acc.y += v.y; acc.z += v.z; acc.w += v.w;
    }
    if (has1) {
        float4 v = *(const float4*)(mF + ((size_t)(b * (S_ - 1) + s - 1)) * H_ + e);
        acc.x += v.x; acc.y += v.y; acc.z += v.z; acc.w += v.w;
    }
    *(float4*)(outF + base) = acc;
}

// -------------------------------------------------------------------------
extern "C" void kernel_launch(void* const* d_in, const int* in_sizes, int n_in,
                              void* d_out, int out_size, void* d_ws, size_t ws_size,
                              hipStream_t stream) {
    const float* seqs = (const float*)d_in[0];
    const int*   d    = (const int*)d_in[1];
    const int*   c    = (const int*)d_in[2];
    const int*   u    = (const int*)d_in[3];
    const float* Ed   = (const float*)d_in[4];
    const float* Ec   = (const float*)d_in[5];
    const float* Eu   = (const float*)d_in[6];
    const float* Wcd  = (const float*)d_in[7];
    const float* Whid = (const float*)d_in[8];
    const float* Wcdh = (const float*)d_in[9];
    float* out = (float*)d_out;

    auto al = [](size_t x) { return (x + 255) & ~(size_t)255; };
    char* w = (char*)d_ws;
    uint16_t* seqsbf = (uint16_t*)w; w += al((size_t)B_ * S_ * H_ * 2);      // 50.3 MB
    uint16_t* A1     = (uint16_t*)w; w += al((size_t)M_ * 256 * 2);          // 16.8 MB
    uint16_t* dcB    = (uint16_t*)w; w += al((size_t)M_ * MIX * 2);          //  8.4 MB
    uint16_t* phB    = (uint16_t*)w; w += al((size_t)M_ * H_ * 2);           // 50.3 MB
    float*    mF     = (float*)w;    w += al((size_t)M_ * H_ * 4);           // 100.6 MB
    uint16_t* WcdB   = (uint16_t*)w; w += al((size_t)MIX * 256 * 2);
    uint16_t* WhidB  = (uint16_t*)w; w += al((size_t)H_ * 2 * H_ * 2);
    uint16_t* WcdhB  = (uint16_t*)w; w += al((size_t)H_ * (MIX + H_) * 2);

    // weights -> bf16 (4 elems/thread)
    {
        int n1 = MIX * (DDIM + CDIM);     // 32768
        int n2 = H_ * 2 * H_;             // 1179648
        int n3 = H_ * (MIX + H_);         // 688128
        cvt_bf16_kernel<<<(n1 / 4 + 255) / 256, 256, 0, stream>>>(Wcd,  WcdB,  n1);
        cvt_bf16_kernel<<<(n2 / 4 + 255) / 256, 256, 0, stream>>>(Whid, WhidB, n2);
        cvt_bf16_kernel<<<(n3 / 4 + 255) / 256, 256, 0, stream>>>(Wcdh, WcdhB, n3);
    }

    // seqs_u (f32 to d_out, bf16 to ws)
    embed_add_kernel<<<B_ * S_, H_ / 4, 0, stream>>>(seqs, u, Eu, out, seqsbf);

    // gather + concat d/c embeddings to bf16
    gather_dc_kernel<<<M_ / 4, 256, 0, stream>>>(d, c, Ed, Ec, A1);

    // GEMM1: d_c = relu(A1 @ W_cd^T)   [M,256]x[256,128]
    {
        constexpr int tasks = 1024 * (128 / 64);
        gemm_relu_wmma<0, 256, 128, false>
            <<<(tasks + 7) / 8, 256, 0, stream>>>(A1, (const uint16_t*)nullptr,
                                                  WcdB, (float*)nullptr, dcB);
    }
    // GEMM2: ph = relu(pairs @ W_hid^T)  [M,1536]x[1536,768], pair view of seqs_u
    {
        constexpr int tasks = 1024 * (768 / 64);
        gemm_relu_wmma<1, 1536, 768, false>
            <<<(tasks + 7) / 8, 256, 0, stream>>>(seqsbf, (const uint16_t*)nullptr,
                                                  WhidB, (float*)nullptr, phB);
    }
    // GEMM3: m = relu(concat(ph, d_c) @ W_cdh^T)  [M,896]x[896,768], f32 out
    {
        constexpr int tasks = 1024 * (768 / 64);
        gemm_relu_wmma<2, 896, 768, true>
            <<<(tasks + 7) / 8, 256, 0, stream>>>(phB, dcB, WcdhB, mF,
                                                  (uint16_t*)nullptr);
    }

    // shifted scatter-adds into out
    scatter_add_kernel<<<B_ * S_, H_ / 4, 0, stream>>>(mF, out);
}
